// GraphContrastiveRetriever_39977555591550
// MI455X (gfx1250) — matrix operand order repaired and checked by reading, not compiled, not run on hardware
//
#include <hip/hip_runtime.h>
#include <hip/hip_bf16.h>

// ---------------- problem constants ----------------
static constexpr int GN  = 2048;   // graph nodes
static constexpr int GHD = 1024;   // gnn hidden == encoder dim
static constexpr int EDI = 1024;   // encoder dim
static constexpr int BQ  = 1024;   // batch
static constexpr int NHD = 8;      // heads
static constexpr int HDD = 128;    // head dim
static constexpr int NLL = 3;      // layers
static constexpr float TEMP_INV = 20.0f;   // 1/0.05
static constexpr float LAPW = 0.1f;

typedef __attribute__((ext_vector_type(16))) __bf16 v16bf;
typedef __attribute__((ext_vector_type(8)))  float  v8f;
typedef __attribute__((ext_vector_type(4)))  unsigned v4u;
typedef __attribute__((ext_vector_type(8)))  int    v8i;
typedef __attribute__((ext_vector_type(4)))  int    v4i;

union FragU { v16bf v; uint4 q[2]; };

// ---------------- helpers ----------------
__device__ __forceinline__ __bf16 f2bf(float f) {
    union { float f; unsigned u; } v; v.f = f;
    unsigned r = v.u + 0x7FFFu + ((v.u >> 16) & 1u);   // round-to-nearest-even
    unsigned short h = (unsigned short)(r >> 16);
    return __builtin_bit_cast(__bf16, h);
}

__device__ __forceinline__ float wave_reduce_sum(float v) {
    #pragma unroll
    for (int off = 16; off > 0; off >>= 1) v += __shfl_xor(v, off, 32);
    return v;
}
__device__ __forceinline__ float wave_reduce_max(float v) {
    #pragma unroll
    for (int off = 16; off > 0; off >>= 1) v = fmaxf(v, __shfl_xor(v, off, 32));
    return v;
}
__device__ __forceinline__ float block_reduce_sum(float v, float* smem) {
    const int lane = threadIdx.x & 31, wid = threadIdx.x >> 5;
    v = wave_reduce_sum(v);
    if (lane == 0) smem[wid] = v;
    __syncthreads();
    float r = (threadIdx.x < (blockDim.x >> 5)) ? smem[threadIdx.x] : 0.f;
    if (wid == 0) { r = wave_reduce_sum(r); if (lane == 0) smem[0] = r; }
    __syncthreads();
    r = smem[0];
    __syncthreads();
    return r;
}
__device__ __forceinline__ float block_reduce_max(float v, float* smem) {
    const int lane = threadIdx.x & 31, wid = threadIdx.x >> 5;
    v = wave_reduce_max(v);
    if (lane == 0) smem[wid] = v;
    __syncthreads();
    float r = (threadIdx.x < (blockDim.x >> 5)) ? smem[threadIdx.x] : -3.0e38f;
    if (wid == 0) { r = wave_reduce_max(r); if (lane == 0) smem[0] = r; }
    __syncthreads();
    r = smem[0];
    __syncthreads();
    return r;
}

// ---------------- TDM: 2-D tile (tile_d1 rows x tile_d0 elems) global -> LDS ----------------
// D# layout per CDNA5 ISA (group0 128b, group1 256b), data_size = 2B.
// clang-23 toolchain: 6-arg builtin (g0 v4u, g1 v8i, g2 v4i, g3 v4i, g4 v8i, cpol i32).
__device__ __forceinline__ void tdm_load_2d(
    unsigned lds_off, const __bf16* gptr,
    unsigned tensor_d0, unsigned tensor_d1,
    unsigned tile_d0, unsigned tile_d1, unsigned stride0_elems)
{
    const unsigned long long ga = (unsigned long long)(size_t)gptr;
    v4u g0;
    g0[0] = 1u;                                            // count=1, user mode
    g0[1] = lds_off;                                       // lds_addr [63:32]
    g0[2] = (unsigned)(ga & 0xffffffffu);                  // global_addr [95:64]
    g0[3] = (unsigned)((ga >> 32) & 0x1ffffffu)            // global_addr [120:96]
          | (2u << 30);                                    // type=2 ("image") [127:126]
    v8i g1;
    g1[0] = (int)(1u << 16);                               // data_size=1 (2 bytes); mask=0
    g1[1] = (int)((tensor_d0 & 0xffffu) << 16);            // tensor_dim0[15:0] @ [63:48]
    g1[2] = (int)((tensor_d0 >> 16) |
                  ((tensor_d1 & 0xffffu) << 16));          // dim0[31:16] | dim1[15:0]
    g1[3] = (int)((tensor_d1 >> 16) | (tile_d0 << 16));    // dim1[31:16] | tile_dim0
    g1[4] = (int)tile_d1;                                  // tile_dim1 | tile_dim2=0
    g1[5] = (int)stride0_elems;                            // tensor_dim0_stride[31:0]
    g1[6] = 0;                                             // stride0[47:32] | stride1[15:0]
    g1[7] = 0;                                             // stride1[47:16]
    const v4i z4 = {0, 0, 0, 0};
    const v8i z8 = {0, 0, 0, 0, 0, 0, 0, 0};
    __builtin_amdgcn_tensor_load_to_lds(g0, g1, z4, z4, z8, 0);
}

// ---------------- TDM-fed LDS-tiled bf16 WMMA GEMM ----------------
// C[M,N] = alpha * A[M,K] @ BT[N,K]^T + bias.  Block: 256 thr (8 waves), 128x128 tile.
// Double-buffered 128x32 A and BT tiles in LDS via tensor_load_to_lds (TENSORcnt).
__global__ __launch_bounds__(256) void gemm_bf16_kernel(
    const __bf16* __restrict__ A, const __bf16* __restrict__ BT,
    const float* __restrict__ bias, float* __restrict__ C,
    int M, int Nn, int K, float alpha)
{
    __shared__ alignas(16) __bf16 As[2][128 * 32];
    __shared__ alignas(16) __bf16 Bs[2][128 * 32];

    const int lane = threadIdx.x & 31;
    const int wid  = threadIdx.x >> 5;
    const int rowB = blockIdx.y * 128;
    const int colB = blockIdx.x * 128;

    const int am    = lane & 15;          // A/C: M within 16-row tile
    const int akb   = (lane >> 4) * 8;    // A: K group base (elems)
    const int bn    = lane & 15;          // B: N within 16-col tile
    const int bkoff = (lane >> 4) * 8;    // B: K group base (elems)
    const int wrow  = wid * 16 + am;      // row within 128-row block

    v8f acc[8] = {{0.f},{0.f},{0.f},{0.f},{0.f},{0.f},{0.f},{0.f}};

    const int T = K / 32;
    if (wid == 0) {
        tdm_load_2d((unsigned)(size_t)&As[0][0], A + (size_t)rowB * K,
                    (unsigned)K, (unsigned)M, 32u, 128u, (unsigned)K);
        tdm_load_2d((unsigned)(size_t)&Bs[0][0], BT + (size_t)colB * K,
                    (unsigned)K, (unsigned)Nn, 32u, 128u, (unsigned)K);
    }

    for (int t = 0; t < T; ++t) {
        const int cur = t & 1;
        if (wid == 0) __builtin_amdgcn_s_wait_tensorcnt(0);
        __syncthreads();                       // tile t resident for all waves
        if (t + 1 < T && wid == 0) {           // prefetch tile t+1 into other buffer
            const int kb = (t + 1) * 32;
            tdm_load_2d((unsigned)(size_t)&As[cur ^ 1][0], A + (size_t)rowB * K + kb,
                        (unsigned)K, (unsigned)M, 32u, 128u, (unsigned)K);
            tdm_load_2d((unsigned)(size_t)&Bs[cur ^ 1][0], BT + (size_t)colB * K + kb,
                        (unsigned)K, (unsigned)Nn, 32u, 128u, (unsigned)K);
        }
        const __bf16* __restrict__ Asb = &As[cur][0];
        const __bf16* __restrict__ Bsb = &Bs[cur][0];

        FragU a;                                // two aligned b128 LDS reads
        a.q[0] = *(const uint4*)(Asb + wrow * 32 + akb);
        a.q[1] = *(const uint4*)(Asb + wrow * 32 + 16 + akb);
        #pragma unroll
        for (int tt = 0; tt < 8; ++tt) {
            const int brow = tt * 16 + bn;
            FragU b;
            b.q[0] = *(const uint4*)(Bsb + brow * 32 + bkoff);
            b.q[1] = *(const uint4*)(Bsb + brow * 32 + bkoff + 16);
            acc[tt] = __builtin_amdgcn_wmma_f32_16x16x32_bf16(
                false, a.v, false, b.v, (short)0, acc[tt], false, false);
        }
        __syncthreads();                       // done reading buf before next overwrite
    }

    const int mo = (lane >> 4) * 8;            // C: M offset from lane group
    #pragma unroll
    for (int tt = 0; tt < 8; ++tt) {
        const int c = colB + tt * 16 + bn;
        const float bv = bias ? bias[c] : 0.f;
        #pragma unroll
        for (int r = 0; r < 8; ++r)
            C[(size_t)(rowB + wid * 16 + mo + r) * Nn + c] = acc[tt][r] * alpha + bv;
    }
}

// ---------------- GAT attention: Hout = att @ Wh, A-tile built on the fly ----------------
// att[i,j] = adj[i,j]>0 ? exp(lrelu(s1_i+s2_j) - M_i)/D_i : 0  (per head)
// WhT bf16 [GHD][GN] (transposed) -> contiguous b128 B-fragment loads.
__global__ __launch_bounds__(256) void attn_gemm_kernel(
    const float* __restrict__ s1, const float* __restrict__ s2,
    const float* __restrict__ Mr, const float* __restrict__ Dr,
    const float* __restrict__ adj, const __bf16* __restrict__ WhT,
    float* __restrict__ Hout)
{
    const int lane = threadIdx.x & 31;
    const int wid  = threadIdx.x >> 5;
    const int row0 = blockIdx.y * 128 + wid * 16;
    const int col0 = blockIdx.x * 64;
    const int head = col0 >> 7;           // HDD = 128

    const int am    = lane & 15;
    const int akb   = (lane >> 4) * 8;
    const int bn    = lane & 15;
    const int bkoff = (lane >> 4) * 8;

    const int irow = row0 + am;
    const float s1v  = s1[head * GN + irow];
    const float mv   = Mr[head * GN + irow];
    const float dv   = Dr[head * GN + irow];
    const float invD = (dv > 0.f) ? (1.f / dv) : 0.f;
    const float* __restrict__ adjrow = adj + (size_t)irow * GN;
    const float* __restrict__ s2h    = s2 + head * GN;
    const __bf16* __restrict__ bp0 = WhT + (size_t)(col0 + bn) * GN;
    const __bf16* __restrict__ bp1 = bp0 + (size_t)16 * GN;
    const __bf16* __restrict__ bp2 = bp0 + (size_t)32 * GN;
    const __bf16* __restrict__ bp3 = bp0 + (size_t)48 * GN;

    v8f acc[4] = {{0.f},{0.f},{0.f},{0.f}};

    for (int kb = 0; kb < GN; kb += 32) {
        FragU a;
        #pragma unroll
        for (int v = 0; v < 8; ++v) {
            const int k0 = kb + ((v < 4) ? (akb + 2 * v) : (16 + akb + 2 * (v - 4)));
            #pragma unroll
            for (int u = 0; u < 2; ++u) {
                const int k = k0 + u;
                float e = s1v + s2h[k];
                e = (e > 0.f) ? e : 0.2f * e;                    // leaky_relu 0.2
                const float p = (adjrow[k] > 0.f) ? __expf(e - mv) * invD : 0.f;
                a.v[2 * v + u] = f2bf(p);
            }
        }
        #pragma unroll
        for (int t = 0; t < 4; ++t) {
            const __bf16* bp = (t == 0) ? bp0 : (t == 1) ? bp1 : (t == 2) ? bp2 : bp3;
            FragU b;
            b.q[0] = *(const uint4*)(bp + kb + bkoff);
            b.q[1] = *(const uint4*)(bp + kb + bkoff + 16);
            acc[t] = __builtin_amdgcn_wmma_f32_16x16x32_bf16(
                false, a.v, false, b.v, (short)0, acc[t], false, false);
        }
    }

    const int mo = (lane >> 4) * 8;
    #pragma unroll
    for (int t = 0; t < 4; ++t) {
        const int c = col0 + t * 16 + bn;
        #pragma unroll
        for (int r = 0; r < 8; ++r)
            Hout[(size_t)(row0 + mo + r) * GHD + c] = acc[t][r];
    }
}

// ---------------- s1/s2 = Wh @ a1, Wh @ a2 (one wave per (h,n)) ----------------
__global__ __launch_bounds__(256) void s1s2_kernel(
    const float* __restrict__ Wh, const float* __restrict__ a1,
    const float* __restrict__ a2, float* __restrict__ s1, float* __restrict__ s2)
{
    const int idx  = blockIdx.x * 8 + (threadIdx.x >> 5);
    const int lane = threadIdx.x & 31;
    if (idx >= NHD * GN) return;
    const int h = idx / GN, n = idx % GN;
    const float* __restrict__ wrow = Wh + (size_t)n * GHD + h * HDD;
    const float* __restrict__ a1h  = a1 + h * HDD;
    const float* __restrict__ a2h  = a2 + h * HDD;
    float t1 = 0.f, t2 = 0.f;
    for (int o = lane; o < HDD; o += 32) {
        const float w = wrow[o];
        t1 += w * a1h[o];
        t2 += w * a2h[o];
    }
    t1 = wave_reduce_sum(t1);
    t2 = wave_reduce_sum(t2);
    if (lane == 0) { s1[idx] = t1; s2[idx] = t2; }
}

// ---------------- online softmax row stats (max + denom) over masked logits ----------------
__global__ __launch_bounds__(256) void rowstats_kernel(
    const float* __restrict__ s1, const float* __restrict__ s2,
    const float* __restrict__ adj, float* __restrict__ Mr, float* __restrict__ Dr)
{
    const int idx  = blockIdx.x * 8 + (threadIdx.x >> 5);
    const int lane = threadIdx.x & 31;
    if (idx >= NHD * GN) return;
    const int h = idx / GN, i = idx % GN;
    const float s1v = s1[h * GN + i];
    const float* __restrict__ adjrow = adj + (size_t)i * GN;
    const float* __restrict__ s2h    = s2 + h * GN;
    float m = -3.0e38f, s = 0.f;
    for (int j = lane; j < GN; j += 32) {
        if (adjrow[j] > 0.f) {
            float e = s1v + s2h[j];
            e = (e > 0.f) ? e : 0.2f * e;
            if (e > m) { s = s * __expf(m - e) + 1.f; m = e; }
            else       { s += __expf(e - m); }
        }
    }
    #pragma unroll
    for (int off = 16; off > 0; off >>= 1) {
        const float m2 = __shfl_xor(m, off, 32);
        const float q2 = __shfl_xor(s, off, 32);
        const float mo = fmaxf(m, m2);
        s = s * __expf(m - mo) + q2 * __expf(m2 - mo);
        m = mo;
    }
    if (lane == 0) { Mr[idx] = m; Dr[idx] = s; }
}

// ---------------- LN -> ELU -> residual (GNN); writes f32 + bf16 ----------------
__global__ __launch_bounds__(256) void ln_elu_res_kernel(
    const float* __restrict__ H, const float* __restrict__ g, const float* __restrict__ b,
    float* __restrict__ X, __bf16* __restrict__ Xbf)
{
    __shared__ float sred[8];
    const int row = blockIdx.x;
    const float* __restrict__ hrow = H + (size_t)row * GHD;
    float v[4]; float sum = 0.f;
    #pragma unroll
    for (int i = 0; i < 4; ++i) { v[i] = hrow[threadIdx.x + i * 256]; sum += v[i]; }
    const float mu = block_reduce_sum(sum, sred) * (1.f / GHD);
    float var = 0.f;
    #pragma unroll
    for (int i = 0; i < 4; ++i) { const float d = v[i] - mu; var += d * d; }
    var = block_reduce_sum(var, sred) * (1.f / GHD);
    const float inv = rsqrtf(var + 1e-5f);
    #pragma unroll
    for (int i = 0; i < 4; ++i) {
        const int c = threadIdx.x + i * 256;
        const float hn = (v[i] - mu) * inv * g[c] + b[c];
        const float e  = (hn > 0.f) ? hn : (__expf(hn) - 1.f);   // elu
        const float xn = X[(size_t)row * GHD + c] + e;           // residual
        X[(size_t)row * GHD + c]   = xn;
        Xbf[(size_t)row * GHD + c] = f2bf(xn);
    }
}

// ---------------- LN -> exact GELU (MLP hidden); writes bf16 ----------------
__global__ __launch_bounds__(256) void ln_gelu_kernel(
    const float* __restrict__ H, const float* __restrict__ g, const float* __restrict__ b,
    __bf16* __restrict__ Obf)
{
    __shared__ float sred[8];
    const int row = blockIdx.x;
    const float* __restrict__ hrow = H + (size_t)row * EDI;
    float v[4]; float sum = 0.f;
    #pragma unroll
    for (int i = 0; i < 4; ++i) { v[i] = hrow[threadIdx.x + i * 256]; sum += v[i]; }
    const float mu = block_reduce_sum(sum, sred) * (1.f / EDI);
    float var = 0.f;
    #pragma unroll
    for (int i = 0; i < 4; ++i) { const float d = v[i] - mu; var += d * d; }
    var = block_reduce_sum(var, sred) * (1.f / EDI);
    const float inv = rsqrtf(var + 1e-5f);
    #pragma unroll
    for (int i = 0; i < 4; ++i) {
        const int c = threadIdx.x + i * 256;
        const float hn = (v[i] - mu) * inv * g[c] + b[c];
        const float ge = 0.5f * hn * (1.f + erff(hn * 0.70710678118f));
        Obf[(size_t)row * EDI + c] = f2bf(ge);
    }
}

// ---------------- row-wise l2 normalize; writes bf16 ----------------
__global__ __launch_bounds__(256) void l2norm_kernel(
    const float* __restrict__ X, __bf16* __restrict__ Obf)
{
    __shared__ float sred[8];
    const int row = blockIdx.x;
    float v[4]; float ss = 0.f;
    #pragma unroll
    for (int i = 0; i < 4; ++i) {
        v[i] = X[(size_t)row * EDI + threadIdx.x + i * 256];
        ss += v[i] * v[i];
    }
    ss = block_reduce_sum(ss, sred);
    const float inv = 1.f / fmaxf(sqrtf(ss), 1e-12f);
    #pragma unroll
    for (int i = 0; i < 4; ++i)
        Obf[(size_t)row * EDI + threadIdx.x + i * 256] = f2bf(v[i] * inv);
}

// ---------------- lap = sum(G * T)/GN via atomics ----------------
__global__ __launch_bounds__(256) void dotlap_kernel(
    const float* __restrict__ G, const float* __restrict__ T, float* __restrict__ out)
{
    __shared__ float sred[8];
    const size_t n = (size_t)GN * EDI;
    float s = 0.f;
    for (size_t i = (size_t)blockIdx.x * blockDim.x + threadIdx.x; i < n;
         i += (size_t)gridDim.x * blockDim.x)
        s += G[i] * T[i];
    s = block_reduce_sum(s, sred);
    if (threadIdx.x == 0) atomicAdd(out + 2, s * (1.f / GN));
}

// ---------------- InfoNCE: per-row logsumexp over [diag, row] ----------------
__global__ __launch_bounds__(256) void lse_kernel(
    const float* __restrict__ S, float* __restrict__ out)
{
    __shared__ float sred[8];
    const int row = blockIdx.x;
    const float* __restrict__ srow = S + (size_t)row * BQ;
    const float diag = srow[row];
    float m = diag;
    for (int j = threadIdx.x; j < BQ; j += 256) m = fmaxf(m, srow[j]);
    m = block_reduce_max(m, sred);
    float s = 0.f;
    for (int j = threadIdx.x; j < BQ; j += 256) s += __expf(srow[j] - m);
    s = block_reduce_sum(s, sred);
    if (threadIdx.x == 0) {
        const float lse = m + logf(s + __expf(diag - m));   // + pos duplicate
        atomicAdd(out + 1, (lse - diag) * (1.f / BQ));
    }
}

// ---------------- converters ----------------
__global__ __launch_bounds__(256) void f2bf_kernel(
    const float* __restrict__ src, __bf16* __restrict__ dst, long long n)
{
    for (long long i = (long long)blockIdx.x * blockDim.x + threadIdx.x; i < n;
         i += (long long)gridDim.x * blockDim.x)
        dst[i] = f2bf(src[i]);
}

// transpose-convert: src f32 [R,Cc] row-major -> dst bf16 [Cc,R]
__global__ __launch_bounds__(256) void f2bfT_kernel(
    const float* __restrict__ src, __bf16* __restrict__ dst, int R, int Cc)
{
    const int n = R * Cc;
    for (int j = blockIdx.x * blockDim.x + threadIdx.x; j < n;
         j += gridDim.x * blockDim.x) {
        const int c = j / R, r = j - c * R;
        dst[j] = f2bf(src[(size_t)r * Cc + c]);
    }
}

// gatW[l] is [NH, ED, HD]; produce W^T bf16 [GHD, EDI]: wt[c][f] = gatW[h=c/HD][f][o=c%HD]
__global__ __launch_bounds__(256) void conv_gatwT_kernel(
    const float* __restrict__ g, __bf16* __restrict__ wt)
{
    const int total = GHD * EDI;
    for (int j = blockIdx.x * blockDim.x + threadIdx.x; j < total;
         j += gridDim.x * blockDim.x) {
        const int c = j >> 10;          // / EDI
        const int f = j & 1023;
        const int h = c >> 7, o = c & 127;
        wt[j] = f2bf(g[((size_t)h * EDI + f) * HDD + o]);
    }
}

__global__ void zero3_kernel(float* out) { if (threadIdx.x < 3) out[threadIdx.x] = 0.f; }
__global__ void finalize_kernel(float* out) { out[0] = out[1] + LAPW * out[2]; }

// ---------------- host orchestration ----------------
extern "C" void kernel_launch(void* const* d_in, const int* in_sizes, int n_in,
                              void* d_out, int out_size, void* d_ws, size_t ws_size,
                              hipStream_t stream) {
    (void)in_sizes; (void)n_in; (void)out_size; (void)ws_size;
    const float* query    = (const float*)d_in[0];
    const float* posdoc   = (const float*)d_in[1];
    const float* nodes    = (const float*)d_in[2];
    const float* adj      = (const float*)d_in[3];
    const float* lapl     = (const float*)d_in[4];
    const float* qW1 = (const float*)d_in[5],  *qb1 = (const float*)d_in[6];
    const float* qg  = (const float*)d_in[7],  *qbt = (const float*)d_in[8];
    const float* qW2 = (const float*)d_in[9],  *qb2 = (const float*)d_in[10];
    const float* dW1 = (const float*)d_in[11], *db1 = (const float*)d_in[12];
    const float* dg  = (const float*)d_in[13], *dbt = (const float*)d_in[14];
    const float* dW2 = (const float*)d_in[15], *db2 = (const float*)d_in[16];
    const float* gatW = (const float*)d_in[17];
    const float* a1   = (const float*)d_in[18];
    const float* a2   = (const float*)d_in[19];
    const float* ln_g = (const float*)d_in[20];
    const float* ln_b = (const float*)d_in[21];
    const float* poW  = (const float*)d_in[22];
    const float* pob  = (const float*)d_in[23];
    float* out = (float*)d_out;

    // ---- workspace layout (256B aligned, with reuse) ----
    char* ws = (char*)d_ws;
    size_t off = 0;
    auto alloc = [&](size_t bytes) -> void* {
        void* p = ws + off;
        off += (bytes + 255) & ~(size_t)255;
        return p;
    };
    float*  XF    = (float*) alloc((size_t)GN * GHD * 4);
    __bf16* XBF   = (__bf16*)alloc((size_t)GN * GHD * 2);
    float*  WH    = (float*) alloc((size_t)GN * GHD * 4);   // reused: LAPBF, QH
    __bf16* WHBT  = (__bf16*)alloc((size_t)GN * GHD * 2);   // Wh^T [GHD][GN]
    float*  HOUT  = (float*) alloc((size_t)GN * GHD * 4);   // reused: LT
    float*  S1    = (float*) alloc((size_t)NHD * GN * 4);
    float*  S2    = (float*) alloc((size_t)NHD * GN * 4);
    float*  MR    = (float*) alloc((size_t)NHD * GN * 4);
    float*  DR    = (float*) alloc((size_t)NHD * GN * 4);
    __bf16* WBT   = (__bf16*)alloc((size_t)EDI * GHD * 2);  // current weight^T (bf16)
    float*  GE    = (float*) alloc((size_t)GN * EDI * 4);   // reused: SMAT
    __bf16* GEBT  = (__bf16*)alloc((size_t)GN * EDI * 2);   // g^T [EDI][GN]
    __bf16* QEBF  = (__bf16*)alloc((size_t)BQ * EDI * 2);
    __bf16* QHBF  = (__bf16*)alloc((size_t)BQ * EDI * 2);
    __bf16* QN    = (__bf16*)alloc((size_t)BQ * EDI * 2);
    __bf16* PN    = (__bf16*)alloc((size_t)BQ * EDI * 2);
    __bf16* LAPBF = (__bf16*)WH;     // [GN,GN] bf16 == 8MB, after GNN
    float*  LT    = HOUT;            // laplacian @ g
    float*  QH    = WH;              // MLP hidden f32
    float*  SMAT  = GE;              // similarity matrix

    const dim3 blk(256);
    const dim3 gemmG (GHD / 128, GN / 128);   // 2048-row GEMMs: (8,16)
    const dim3 gemmGB(EDI / 128, BQ / 128);   // 1024-row GEMMs: (8,8)
    const dim3 attnG (GHD / 64, GN / 128);    // (16,16)
    const int  rowsHN = (NHD * GN) / 8;       // 8 waves/block

    zero3_kernel<<<1, 32, 0, stream>>>(out);

    // ---- GNN ----
    (void)hipMemcpyAsync(XF, nodes, (size_t)GN * GHD * 4, hipMemcpyDeviceToDevice, stream);
    f2bf_kernel<<<4096, blk, 0, stream>>>(nodes, XBF, (long long)GN * GHD);

    for (int l = 0; l < NLL; ++l) {
        conv_gatwT_kernel<<<4096, blk, 0, stream>>>(gatW + (size_t)l * NHD * EDI * HDD, WBT);
        gemm_bf16_kernel<<<gemmG, blk, 0, stream>>>(XBF, WBT, nullptr, WH,
                                                    GN, GHD, EDI, 1.0f);
        f2bfT_kernel<<<4096, blk, 0, stream>>>(WH, WHBT, GN, GHD);   // -> [GHD][GN]
        s1s2_kernel<<<rowsHN, blk, 0, stream>>>(WH, a1 + l * NHD * HDD,
                                                a2 + l * NHD * HDD, S1, S2);
        rowstats_kernel<<<rowsHN, blk, 0, stream>>>(S1, S2, adj, MR, DR);
        attn_gemm_kernel<<<attnG, blk, 0, stream>>>(S1, S2, MR, DR, adj, WHBT, HOUT);
        ln_elu_res_kernel<<<GN, blk, 0, stream>>>(HOUT, ln_g + l * GHD, ln_b + l * GHD,
                                                  XF, XBF);
    }

    // graph_embeds = x @ poW + pob
    f2bfT_kernel<<<4096, blk, 0, stream>>>(poW, WBT, GHD, EDI);      // -> [EDI][GHD]
    gemm_bf16_kernel<<<gemmG, blk, 0, stream>>>(XBF, WBT, pob, GE, GN, EDI, GHD, 1.0f);
    f2bfT_kernel<<<4096, blk, 0, stream>>>(GE, GEBT, GN, EDI);       // -> [EDI][GN]

    // lap = sum(g * (L @ g)) / GN
    f2bf_kernel<<<8192, blk, 0, stream>>>(lapl, LAPBF, (long long)GN * GN);
    gemm_bf16_kernel<<<gemmG, blk, 0, stream>>>(LAPBF, GEBT, nullptr, LT,
                                                GN, EDI, GN, 1.0f);
    dotlap_kernel<<<2048, blk, 0, stream>>>(GE, LT, out);

    // ---- query MLP ----
    f2bf_kernel<<<4096, blk, 0, stream>>>(query, QEBF, (long long)BQ * EDI);
    f2bfT_kernel<<<4096, blk, 0, stream>>>(qW1, WBT, EDI, EDI);
    gemm_bf16_kernel<<<gemmGB, blk, 0, stream>>>(QEBF, WBT, qb1, QH, BQ, EDI, EDI, 1.0f);
    ln_gelu_kernel<<<BQ, blk, 0, stream>>>(QH, qg, qbt, QHBF);
    f2bfT_kernel<<<4096, blk, 0, stream>>>(qW2, WBT, EDI, EDI);
    gemm_bf16_kernel<<<gemmGB, blk, 0, stream>>>(QHBF, WBT, qb2, QH, BQ, EDI, EDI, 1.0f);
    l2norm_kernel<<<BQ, blk, 0, stream>>>(QH, QN);

    // ---- doc MLP ----
    f2bf_kernel<<<4096, blk, 0, stream>>>(posdoc, QEBF, (long long)BQ * EDI);
    f2bfT_kernel<<<4096, blk, 0, stream>>>(dW1, WBT, EDI, EDI);
    gemm_bf16_kernel<<<gemmGB, blk, 0, stream>>>(QEBF, WBT, db1, QH, BQ, EDI, EDI, 1.0f);
    ln_gelu_kernel<<<BQ, blk, 0, stream>>>(QH, dg, dbt, QHBF);
    f2bfT_kernel<<<4096, blk, 0, stream>>>(dW2, WBT, EDI, EDI);
    gemm_bf16_kernel<<<gemmGB, blk, 0, stream>>>(QHBF, WBT, db2, QH, BQ, EDI, EDI, 1.0f);
    l2norm_kernel<<<BQ, blk, 0, stream>>>(QH, PN);

    // ---- similarity + InfoNCE (B operand PN is naturally [N,K]) ----
    gemm_bf16_kernel<<<gemmGB, blk, 0, stream>>>(QN, PN, nullptr, SMAT,
                                                 BQ, BQ, EDI, TEMP_INV);
    lse_kernel<<<BQ, blk, 0, stream>>>(SMAT, out);

    finalize_kernel<<<1, 1, 0, stream>>>(out);
}
